// BranchingLayer_52707838657334
// MI455X (gfx1250) — compile-verified, matrix operand order checked by reading, not run on hardware
//
#include <hip/hip_runtime.h>
#include <hip/hip_bf16.h>

typedef __attribute__((ext_vector_type(16))) _Float16 v16h;
typedef __attribute__((ext_vector_type(8)))  _Float16 v8h;
typedef __attribute__((ext_vector_type(8)))  float    v8f;

#define BATCH     2048
#define NPAR      256
#define NFEAT     32
#define NGLOB     16
#define NBR       4
#define ROW_OFF   262144
#define ROWS_X    786432            // ROW_OFF + NPAR*BATCH
#define PB        524288            // NPAR*BATCH
#define DOUT      128

// ---------------------------------------------------------------------------
// Kernel 0: pre-pack W1 (48x128, zero-padded to 64x128) and W2 (128x128) into
// per-lane WMMA B-fragment order (f16). Fragment f = s*8 + t is 32 lanes x 16
// halves; element (l,e) maps to k = lo+e (e<8) or 16+lo+(e-8), lo = 8*(l>=16).
// ---------------------------------------------------------------------------
__global__ void prepack_weights(const float* __restrict__ W1,
                                const float* __restrict__ W2,
                                _Float16* __restrict__ W1p,   // 16 frags = 8192 halves
                                _Float16* __restrict__ W2p) { // 32 frags = 16384 halves
  int i = blockIdx.x * 256 + threadIdx.x;        // 0 .. 24575
  if (i >= 24576) return;
  int j    = (i < 8192) ? i : (i - 8192);
  int frag = j >> 9;
  int l    = (j >> 4) & 31;
  int e    = j & 15;
  int s = frag >> 3, t = frag & 7;
  int lo = (l >> 4) * 8;
  int k  = (e < 8) ? (lo + e) : (16 + lo + (e - 8));
  int K  = 32 * s + k;
  int n  = 16 * t + (l & 15);
  if (i < 8192) {
    float v = (K < 48) ? W1[K * DOUT + n] : 0.0f;
    W1p[j] = (_Float16)v;
  } else {
    W2p[j] = (_Float16)W2[K * DOUT + n];
  }
}

// ---------------------------------------------------------------------------
// Kernel 1: copy x -> head of output (float4 vectorized)
// ---------------------------------------------------------------------------
__global__ void copy_x4(const float4* __restrict__ src, float4* __restrict__ dst, int n4) {
  int i = blockIdx.x * 256 + threadIdx.x;
  if (i < n4) dst[i] = src[i];
}

// ---------------------------------------------------------------------------
// Kernel 2: fused gather + MLP (WMMA f16->f32) + residual + branch permute
// 8 waves / block, 16 rows / wave, grid = PB/128 = 4096 blocks.
// ---------------------------------------------------------------------------
__global__ __launch_bounds__(256) void branch_mlp(
    const float* __restrict__ x, const float* __restrict__ gf,
    const int* __restrict__ pidx, const float* __restrict__ b1,
    const float* __restrict__ b2, const _Float16* __restrict__ W1p,
    const _Float16* __restrict__ W2p, float* __restrict__ out) {

  __shared__ _Float16 hs[8][16][DOUT];   // 32 KB: layer-1 activations (f16)
  __shared__ float    xs[8][16][NFEAT];  // 16 KB: parent features (residual)

  const int tid   = threadIdx.x;
  const int w     = tid >> 5;
  const int l     = tid & 31;
  const int mrow  = l & 15;        // A-layout / staging row this lane loads
  const int khalf = (l >> 4) * 8;  // 0 for lanes 0-15, 8 for lanes 16-31
  const int base  = (blockIdx.x * 8 + w) * 16;   // tile base row in [0, PB)

  // ---- gather A fragments (layer-1 input: [x(32) | g(16) | pad(16)]) ----
  const int   src = pidx[base + mrow];
  const float* xr = x  + (long)src * NFEAT;
  const float* gr = gf + (long)(src & (BATCH - 1)) * NGLOB;

  float fx[16];
#pragma unroll
  for (int e = 0; e < 8; ++e) fx[e]     = xr[khalf + e];
#pragma unroll
  for (int e = 0; e < 8; ++e) fx[8 + e] = xr[16 + khalf + e];
#pragma unroll
  for (int e = 0; e < 8; ++e) { xs[w][mrow][khalf + e] = fx[e]; xs[w][mrow][16 + khalf + e] = fx[8 + e]; }

  v16h a0, a1;
#pragma unroll
  for (int e = 0; e < 16; ++e) a0[e] = (_Float16)fx[e];
#pragma unroll
  for (int e = 0; e < 8; ++e)  { a1[e] = (_Float16)gr[khalf + e]; a1[8 + e] = (_Float16)0.0f; }

  // ---- layer 1: h1 = leaky_relu([x|g] @ W1 + b1), K = 64 (padded) ----
#pragma unroll
  for (int t = 0; t < 8; ++t) {
    v16h bA = *(const v16h*)(W1p + (0 * 8 + t) * 512 + l * 16);
    v16h bB = *(const v16h*)(W1p + (1 * 8 + t) * 512 + l * 16);
    v8f c = {};
    c = __builtin_amdgcn_wmma_f32_16x16x32_f16(false, a0, false, bA, (short)0, c, false, false);
    c = __builtin_amdgcn_wmma_f32_16x16x32_f16(false, a1, false, bB, (short)0, c, false, false);
    const int n = 16 * t + (l & 15);
    const float bias = b1[n];
#pragma unroll
    for (int v = 0; v < 8; ++v) {
      const int m = v + 8 * (l >> 4);
      float hv = c[v] + bias;
      hv = (hv > 0.0f) ? hv : 0.01f * hv;
      hs[w][m][n] = (_Float16)hv;
    }
  }
  __syncthreads();

  // ---- reload h1 in A-fragment layout (K = 128 -> 4 fragments) ----
  v16h a2[4];
#pragma unroll
  for (int s = 0; s < 4; ++s) {
    v8h lo = *(const v8h*)&hs[w][mrow][32 * s + khalf];
    v8h hi = *(const v8h*)&hs[w][mrow][32 * s + 16 + khalf];
#pragma unroll
    for (int e = 0; e < 8; ++e) { a2[s][e] = lo[e]; a2[s][8 + e] = hi[e]; }
  }

  // ---- layer 2 + bias + residual + branch-permuted scatter ----
  const int pgrp  = base >> 11;         // parent index (constant in tile)
  const int bevt0 = base & (BATCH - 1); // event index of row m=0
#pragma unroll
  for (int t = 0; t < 8; ++t) {
    v8f c = {};
#pragma unroll
    for (int s = 0; s < 4; ++s) {
      v16h bb = *(const v16h*)(W2p + (s * 8 + t) * 512 + l * 16);
      c = __builtin_amdgcn_wmma_f32_16x16x32_f16(false, a2[s], false, bb, (short)0, c, false, false);
    }
    const int n     = 16 * t + (l & 15);
    const float bias = b2[n];
    const int br    = n >> 5;        // branch  = col / 32
    const int fcol  = n & 31;        // feature = col % 32
    const int featr = n >> 2;        // residual feature = col / 4
    const long rowBase = (long)ROWS_X + (long)(pgrp * NBR + br) * BATCH + bevt0;
#pragma unroll
    for (int v = 0; v < 8; ++v) {
      const int m = v + 8 * (l >> 4);
      const float val = c[v] + bias + xs[w][m][featr];
      out[(rowBase + m) * NFEAT + fcol] = val;
    }
  }
}

extern "C" void kernel_launch(void* const* d_in, const int* in_sizes, int n_in,
                              void* d_out, int out_size, void* d_ws, size_t ws_size,
                              hipStream_t stream) {
  const float* x   = (const float*)d_in[0];
  const float* gf  = (const float*)d_in[1];
  const int*  pidx = (const int*)d_in[2];
  const float* W1  = (const float*)d_in[3];
  const float* b1  = (const float*)d_in[4];
  const float* W2  = (const float*)d_in[5];
  const float* b2  = (const float*)d_in[6];
  float* out = (float*)d_out;

  _Float16* W1p = (_Float16*)d_ws;       // 8192 halves (16 KB)
  _Float16* W2p = W1p + 8192;            // 16384 halves (32 KB)

  prepack_weights<<<96, 256, 0, stream>>>(W1, W2, W1p, W2p);

  const int n4 = in_sizes[0] / 4;        // x is 786432*32 floats
  copy_x4<<<(n4 + 255) / 256, 256, 0, stream>>>((const float4*)x, (float4*)out, n4);

  branch_mlp<<<PB / 128, 256, 0, stream>>>(x, gf, pidx, b1, b2, W1p, W2p, out);
}
